// LinearHelixTransformerBlock_26637387169981
// MI455X (gfx1250) — matrix operand order repaired and checked by reading, not compiled, not run on hardware
//
#include <hip/hip_runtime.h>
#include <hip/hip_bf16.h>
#include <stdint.h>

#define DIM   384
#define HEADS 6
#define HD    64
#define NTOK  1024
#define BATCH 16
#define BN    (BATCH * NTOK)   // 16384 rows
#define MLPH  1536
#define ATT_SCALE 0.125f       // 1/sqrt(64)

typedef __attribute__((ext_vector_type(16))) _Float16 v16h;
typedef __attribute__((ext_vector_type(8)))  float    v8f;
typedef __attribute__((ext_vector_type(8)))  _Float16 h8v;

// ---- CDNA5 wave32 WMMA fragment helpers (contiguous k order: k = 16*(lane/16) + s, same for A and B) ----

// A-matrix 16x32 f16: lane L -> row = L%16; VGPR v holds K pair:
//   K = 2v + 16*(L/16) (same k order as B)
__device__ __forceinline__ int kfA(int v, int half) {
  return 2 * v + 16 * half;
}
// B-matrix 32x16 f16: lane L -> col n = L%16; lanes 0-15 hold K=0..15, 16-31 K=16..31
__device__ __forceinline__ int kfB(int v, int half) { return 2 * v + 16 * half; }

// A fragment from row-major [row][k] LDS, row stride ld
__device__ __forceinline__ v16h fragA(const _Float16* base, int ld) {
  int lane = threadIdx.x & 31;
  int row = lane & 15, half = lane >> 4;
  v16h f;
#pragma unroll
  for (int v = 0; v < 8; ++v) {
    int k = kfA(v, half);
    f[2 * v]     = base[row * ld + k];
    f[2 * v + 1] = base[row * ld + k + 1];
  }
  return f;
}

// B fragment, source stored K-major: B[k][n] at base[k*ld + n]
__device__ __forceinline__ v16h fragB_kmajor(const _Float16* base, int ld) {
  int lane = threadIdx.x & 31;
  int n = lane & 15, half = lane >> 4;
  v16h f;
#pragma unroll
  for (int v = 0; v < 8; ++v) {
    int k = kfB(v, half);
    f[2 * v]     = base[k * ld + n];
    f[2 * v + 1] = base[(k + 1) * ld + n];
  }
  return f;
}

// B fragment where B[k][n] = S[n][k] (S row-major, stride ld) -- e.g. scores = Q @ K^T
__device__ __forceinline__ v16h fragB_nmajor(const _Float16* base, int ld) {
  int lane = threadIdx.x & 31;
  int n = lane & 15, half = lane >> 4;
  v16h f;
#pragma unroll
  for (int v = 0; v < 8; ++v) {
    int k = kfB(v, half);
    f[2 * v]     = base[n * ld + k];
    f[2 * v + 1] = base[n * ld + k + 1];
  }
  return f;
}

__device__ __forceinline__ v8f wmma_f16(v16h a, v16h b, v8f c) {
  return __builtin_amdgcn_wmma_f32_16x16x32_f16(false, a, false, b, (short)0, c,
                                                false, false);
}

// ---- CDNA5 async copy engine: global -> LDS, 16B per lane, tracked by ASYNCcnt ----
// LDS address = flat pointer truncated to 32 bits.
__device__ __forceinline__ void async_copy16(const void* g, unsigned lds_off) {
  unsigned long long ga = (unsigned long long)(uintptr_t)g;
  asm volatile("global_load_async_to_lds_b128 %0, %1, off"
               :: "v"(lds_off), "v"(ga) : "memory");
}
__device__ __forceinline__ unsigned lds_addr(const void* p) {
  return (unsigned)(uintptr_t)p;
}

// ---------------- weight fp32 -> f16 ----------------
__global__ void cvt_f32_f16(const float* __restrict__ in, _Float16* __restrict__ out, int n) {
  int i = blockIdx.x * 256 + threadIdx.x;
  if (i < n) out[i] = (_Float16)in[i];
}

// ------------- transpose-gather + LayerNorm: x[B,C,N] -> xn f16 [B,N,C] -------------
__global__ __launch_bounds__(256) void ln_embed_kernel(
    const float* __restrict__ x, const float* __restrict__ w,
    const float* __restrict__ bia, _Float16* __restrict__ xn) {
  __shared__ float tile[DIM][33];  // 32 tokens, padded
  const int b = blockIdx.y, t0 = blockIdx.x * 32;
  for (int i = threadIdx.x; i < DIM * 32; i += 256) {
    int c = i >> 5, tk = i & 31;
    tile[c][tk] = x[((size_t)b * DIM + c) * NTOK + t0 + tk];  // coalesced over tk
  }
  __syncthreads();
  int tok = threadIdx.x >> 3, part = threadIdx.x & 7;  // 8 lanes per token, same wave
  float s = 0.f;
  for (int c = part; c < DIM; c += 8) s += tile[c][tok];
  s += __shfl_xor(s, 1); s += __shfl_xor(s, 2); s += __shfl_xor(s, 4);
  float mu = s * (1.f / DIM);
  float var = 0.f;
  for (int c = part; c < DIM; c += 8) { float d = tile[c][tok] - mu; var += d * d; }
  var += __shfl_xor(var, 1); var += __shfl_xor(var, 2); var += __shfl_xor(var, 4);
  float rstd = rsqrtf(var * (1.f / DIM) + 1e-5f);
  _Float16* dst = xn + ((size_t)b * NTOK + t0 + tok) * DIM;
  for (int c = part; c < DIM; c += 8)
    dst[c] = (_Float16)((tile[c][tok] - mu) * rstd * w[c] + bia[c]);
}

// ------------- LayerNorm over contiguous rows [M,384] f32 -> f16 -------------
__global__ __launch_bounds__(256) void ln_rows_kernel(
    const float* __restrict__ in, const float* __restrict__ w,
    const float* __restrict__ bia, _Float16* __restrict__ out) {
  int row = blockIdx.x * 32 + (threadIdx.x >> 3);
  int part = threadIdx.x & 7;
  const float* r = in + (size_t)row * DIM;
  float s = 0.f;
  for (int c = part; c < DIM; c += 8) s += r[c];
  s += __shfl_xor(s, 1); s += __shfl_xor(s, 2); s += __shfl_xor(s, 4);
  float mu = s * (1.f / DIM);
  float var = 0.f;
  for (int c = part; c < DIM; c += 8) { float d = r[c] - mu; var += d * d; }
  var += __shfl_xor(var, 1); var += __shfl_xor(var, 2); var += __shfl_xor(var, 4);
  float rstd = rsqrtf(var * (1.f / DIM) + 1e-5f);
  _Float16* o = out + (size_t)row * DIM;
  for (int c = part; c < DIM; c += 8)
    o[c] = (_Float16)((r[c] - mu) * rstd * w[c] + bia[c]);
}

// ------------- generic f16 WMMA GEMM, async double-buffered LDS, fused epilogues -------------
// C[M,N] = A[M,K] @ W[K,N], block tile 128x64, K chunks of 32, 8 waves (16 rows each)
// EPI 0: scatter k/q/v f16   EPI 1: +bias +emb(x gather) -> f32 pre
// EPI 2: +bias, GELU -> f16  EPI 3: +bias +residual -> transposed f32 out
template <int EPI>
__global__ __launch_bounds__(256) void gemm16(
    const _Float16* __restrict__ A, const _Float16* __restrict__ W,
    int N, int K,
    const float* __restrict__ bias, const float* __restrict__ aux,
    float* __restrict__ out32, _Float16* __restrict__ out16,
    _Float16* __restrict__ outK, _Float16* __restrict__ outQ,
    _Float16* __restrict__ outV) {
  __shared__ __align__(16) _Float16 aT[2][128][32];  // 16 KB (double buffered)
  __shared__ __align__(16) _Float16 wT[2][32][64];   // 8 KB
  const int m0 = blockIdx.x * 128, n0 = blockIdx.y * 64;
  const int wv = threadIdx.x >> 5;

  // issue one stage of async global->LDS DMA (3 async instructions per wave)
  auto stage = [&](int buf, int k0) {
    for (int L = threadIdx.x; L < 512; L += 256) {  // A tile: 128x32 f16
      int r = L >> 2, s = L & 3;
      async_copy16(&A[(size_t)(m0 + r) * K + k0 + s * 8],
                   lds_addr(&aT[buf][r][s * 8]));
    }
    {  // W tile: 32x64 f16
      int r = threadIdx.x >> 3, s = threadIdx.x & 7;
      async_copy16(&W[(size_t)(k0 + r) * N + n0 + s * 8],
                   lds_addr(&wT[buf][r][s * 8]));
    }
  };

  v8f acc[4];
#pragma unroll
  for (int j = 0; j < 4; ++j)
#pragma unroll
    for (int e = 0; e < 8; ++e) acc[j][e] = 0.f;

  stage(0, 0);
  for (int k0 = 0; k0 < K; k0 += 32) {
    const int buf = (k0 >> 5) & 1;
    if (k0 + 64 < K)  // L2 prefetch for the tile after next
      __builtin_prefetch(&A[(size_t)(m0 + (threadIdx.x >> 1)) * K + k0 + 64], 0, 3);
    __syncthreads();  // all waves done reading buf^1
    if (k0 + 32 < K) {
      stage(buf ^ 1, k0 + 32);  // DMA next tile while we compute this one
      asm volatile("s_wait_asynccnt 0x3" ::: "memory");  // current tile landed
    } else {
      asm volatile("s_wait_asynccnt 0x0" ::: "memory");
    }
    __syncthreads();  // current tile visible to all waves
    v16h af = fragA(&aT[buf][16 * wv][0], 32);
#pragma unroll
    for (int j = 0; j < 4; ++j) {
      v16h bf = fragB_kmajor(&wT[buf][0][16 * j], 64);
      acc[j] = wmma_f16(af, bf, acc[j]);
    }
  }

  const int lane = threadIdx.x & 31, half = lane >> 4, nl = lane & 15;
#pragma unroll
  for (int j = 0; j < 4; ++j)
#pragma unroll
    for (int r = 0; r < 8; ++r) {
      int m = m0 + 16 * wv + r + 8 * half;  // C layout: VGPR r -> M=r / r+8
      int n = n0 + 16 * j + nl;
      float c = acc[j][r];
      int b = m >> 10, tok = m & 1023;
      if constexpr (EPI == 0) {
        int t = n / DIM, rem = n % DIM, head = rem >> 6, hd = rem & 63;
        _Float16* dst = (t == 0) ? outK : (t == 1) ? outQ : outV;
        dst[(((size_t)b * HEADS + head) * NTOK + tok) * HD + hd] = (_Float16)c;
      } else if constexpr (EPI == 1) {
        c += bias[n] + aux[((size_t)b * DIM + n) * NTOK + tok];  // + emb residual
        out32[(size_t)m * DIM + n] = c;
      } else if constexpr (EPI == 2) {
        c += bias[n];
        float g = 0.5f * c * (1.f + erff(c * 0.70710678118f));  // exact GELU
        out16[(size_t)m * N + n] = (_Float16)g;
      } else {
        c += bias[n] + aux[(size_t)m * DIM + n];                // + residual
        out32[((size_t)b * DIM + n) * NTOK + tok] = c;          // transposed out
      }
    }
}

// ------------- flash cross-attention: one (b,head), 128 queries / block -------------
// Row-sum of the probability tile is computed on the matrix unit (P @ ones) instead of
// lane shuffles: every output column of that WMMA holds the row sum, in the same
// C-layout rows as the correction factors, so only the running-max needs ds_bpermute.
__global__ __launch_bounds__(256) void attn_kernel(
    const _Float16* __restrict__ Q, const _Float16* __restrict__ Km,
    const _Float16* __restrict__ Vm, _Float16* __restrict__ O) {
  __shared__ __align__(16) _Float16 kv[2][2][32][64]; // [buf][K/V]: 16 KB
  __shared__ __align__(16) _Float16 qs[128][64];      // 16 KB
  __shared__ __align__(16) _Float16 pst[8][16][32];   // per-wave P staging: 8 KB
  const int bh = blockIdx.y;           // b*HEADS + head
  const int q0 = blockIdx.x * 128;
  const int wv = threadIdx.x >> 5, lane = threadIdx.x & 31;
  const int half = lane >> 4, nl = lane & 15;

  const _Float16* qp = Q + ((size_t)bh * NTOK + q0) * HD;
  const _Float16* kp = Km + (size_t)bh * NTOK * HD;
  const _Float16* vp = Vm + (size_t)bh * NTOK * HD;

  // one K/V stage = 2 async instructions per wave
  auto stageKV = [&](int buf, int kt) {
    for (int L = threadIdx.x; L < 512; L += 256) {
      int which = L >> 8, j = L & 255, r = j >> 3, s = j & 7;
      const _Float16* src = which ? vp : kp;
      async_copy16(&src[(size_t)(kt + r) * HD + s * 8],
                   lds_addr(&kv[buf][which][r][s * 8]));
    }
  };

  // Q tile via async DMA: 4 instructions per wave
  for (int L = threadIdx.x; L < 1024; L += 256) {
    int r = L >> 3, s = L & 7;
    async_copy16(&qp[(size_t)r * HD + s * 8], lds_addr(&qs[r][s * 8]));
  }
  stageKV(0, 0);
  asm volatile("s_wait_asynccnt 0x2" ::: "memory");  // Q landed; KV tile 0 in flight
  __syncthreads();
  v16h qf0 = fragA(&qs[16 * wv][0], 64);    // d = 0..31
  v16h qf1 = fragA(&qs[16 * wv][32], 64);   // d = 32..63

  v16h ones;  // all-ones B fragment: D = P @ 1 -> per-row sums in every lane
#pragma unroll
  for (int i = 0; i < 16; ++i) ones[i] = (_Float16)1.0f;

  float mst[8];
  v8f oacc[4], lacc;
#pragma unroll
  for (int r = 0; r < 8; ++r) { mst[r] = -1e30f; lacc[r] = 0.f; }
#pragma unroll
  for (int j = 0; j < 4; ++j)
#pragma unroll
    for (int e = 0; e < 8; ++e) oacc[j][e] = 0.f;

  for (int kt = 0; kt < NTOK; kt += 32) {
    const int buf = (kt >> 5) & 1;
    __syncthreads();  // all waves done reading buf^1
    if (kt + 32 < NTOK) {
      stageKV(buf ^ 1, kt + 32);  // DMA next K/V tile
      asm volatile("s_wait_asynccnt 0x2" ::: "memory");  // current tile landed
    } else {
      asm volatile("s_wait_asynccnt 0x0" ::: "memory");
    }
    __syncthreads();

    v8f s0, s1;
#pragma unroll
    for (int e = 0; e < 8; ++e) { s0[e] = 0.f; s1[e] = 0.f; }
    {  // scores: 16 queries x 32 keys, contraction over hd=64
      v16h kb0 = fragB_nmajor(&kv[buf][0][0][0], 64);
      v16h kb1 = fragB_nmajor(&kv[buf][0][0][32], 64);
      v16h kc0 = fragB_nmajor(&kv[buf][0][16][0], 64);
      v16h kc1 = fragB_nmajor(&kv[buf][0][16][32], 64);
      s0 = wmma_f16(qf0, kb0, s0);
      s0 = wmma_f16(qf1, kb1, s0);
      s1 = wmma_f16(qf0, kc0, s1);
      s1 = wmma_f16(qf1, kc1, s1);
    }
    float corr[8];
#pragma unroll
    for (int r = 0; r < 8; ++r) {  // online softmax; row = r + 8*half across 16 lanes
      float a = s0[r] * ATT_SCALE, b2 = s1[r] * ATT_SCALE;
      float mx = fmaxf(a, b2);
      mx = fmaxf(mx, __shfl_xor(mx, 1));
      mx = fmaxf(mx, __shfl_xor(mx, 2));
      mx = fmaxf(mx, __shfl_xor(mx, 4));
      mx = fmaxf(mx, __shfl_xor(mx, 8));
      float mn = fmaxf(mst[r], mx);
      corr[r] = __expf(mst[r] - mn);
      mst[r] = mn;
      float e0 = __expf(a - mn), e1 = __expf(b2 - mn);
      pst[wv][r + 8 * half][nl] = (_Float16)e0;        // C-layout -> LDS
      pst[wv][r + 8 * half][16 + nl] = (_Float16)e1;
    }
#pragma unroll
    for (int r = 0; r < 8; ++r) lacc[r] *= corr[r];
#pragma unroll
    for (int j = 0; j < 4; ++j)
#pragma unroll
      for (int r = 0; r < 8; ++r) oacc[j][r] *= corr[r];
    asm volatile("s_wait_dscnt 0x0" ::: "memory");     // wave-private LDS RAW
    v16h pf = fragA(&pst[wv][0][0], 32);               // A-layout reload of P
    lacc = wmma_f16(pf, ones, lacc);                   // row sums on the XDL pipe
#pragma unroll
    for (int j = 0; j < 4; ++j) {
      v16h vb = fragB_kmajor(&kv[buf][1][0][16 * j], 64);  // V[k][hd]
      oacc[j] = wmma_f16(pf, vb, oacc[j]);
    }
  }

  const int b = bh / HEADS, head = bh % HEADS;
#pragma unroll
  for (int j = 0; j < 4; ++j)
#pragma unroll
    for (int r = 0; r < 8; ++r) {
      int tok = q0 + 16 * wv + r + 8 * half;
      int c = head * HD + 16 * j + nl;
      O[((size_t)b * NTOK + tok) * DIM + c] = (_Float16)(oacc[j][r] / lacc[r]);
    }
}

// ---------------- launch ----------------
extern "C" void kernel_launch(void* const* d_in, const int* in_sizes, int n_in,
                              void* d_out, int out_size, void* d_ws, size_t ws_size,
                              hipStream_t stream) {
  (void)in_sizes; (void)n_in; (void)out_size; (void)ws_size;
  const float* x1 = (const float*)d_in[0];
  const float* x2 = (const float*)d_in[1];
  const float* lna1w = (const float*)d_in[2];
  const float* lna1b = (const float*)d_in[3];
  const float* kqv1w = (const float*)d_in[4];
  const float* lna2w = (const float*)d_in[5];
  const float* lna2b = (const float*)d_in[6];
  const float* kqv2w = (const float*)d_in[7];
  const float* proj1w = (const float*)d_in[8];
  const float* proj1b = (const float*)d_in[9];
  const float* proj2w = (const float*)d_in[10];
  const float* proj2b = (const float*)d_in[11];
  const float* ln1w = (const float*)d_in[12];
  const float* ln1b = (const float*)d_in[13];
  const float* ln2w = (const float*)d_in[14];
  const float* ln2b = (const float*)d_in[15];
  const float* fc1w1 = (const float*)d_in[16];
  const float* fc1b1 = (const float*)d_in[17];
  const float* fc2w1 = (const float*)d_in[18];
  const float* fc2b1 = (const float*)d_in[19];
  const float* fc1w2 = (const float*)d_in[20];
  const float* fc1b2 = (const float*)d_in[21];
  const float* fc2w2 = (const float*)d_in[22];
  const float* fc2b2 = (const float*)d_in[23];
  float* out = (float*)d_out;

  uint8_t* ws = (uint8_t*)d_ws;
  size_t cur = 0;
  auto take = [&](size_t bytes) -> void* {
    void* p = ws + cur;
    cur += (bytes + 255) & ~(size_t)255;
    return p;
  };
  const size_t EMB = (size_t)BN * DIM;  // 6,291,456 elements
  _Float16* xn1 = (_Float16*)take(EMB * 2);
  _Float16* xn2 = (_Float16*)take(EMB * 2);
  _Float16* k1 = (_Float16*)take(EMB * 2);
  _Float16* q1 = (_Float16*)take(EMB * 2);
  _Float16* v1 = (_Float16*)take(EMB * 2);
  _Float16* k2 = (_Float16*)take(EMB * 2);
  _Float16* q2 = (_Float16*)take(EMB * 2);
  _Float16* v2 = (_Float16*)take(EMB * 2);
  _Float16* a1 = (_Float16*)take(EMB * 2);
  _Float16* a2 = (_Float16*)take(EMB * 2);
  float* pre1 = (float*)take(EMB * 4);
  float* pre2 = (float*)take(EMB * 4);
  _Float16* wkqv1 = (_Float16*)take((size_t)DIM * 3 * DIM * 2);
  _Float16* wkqv2 = (_Float16*)take((size_t)DIM * 3 * DIM * 2);
  _Float16* wproj1 = (_Float16*)take((size_t)DIM * DIM * 2);
  _Float16* wproj2 = (_Float16*)take((size_t)DIM * DIM * 2);
  _Float16* wfc1_1 = (_Float16*)take((size_t)DIM * MLPH * 2);
  _Float16* wfc2_1 = (_Float16*)take((size_t)MLPH * DIM * 2);
  _Float16* wfc1_2 = (_Float16*)take((size_t)DIM * MLPH * 2);
  _Float16* wfc2_2 = (_Float16*)take((size_t)MLPH * DIM * 2);
  // MLP hidden aliases the (then-dead) q/k/v + attn-out region: 8*EMB*2 == 2*BN*MLPH*2
  _Float16* hid1 = k1;
  _Float16* hid2 = k1 + (size_t)BN * MLPH;

  auto cvt = [&](const float* src, _Float16* dst, int n) {
    cvt_f32_f16<<<(n + 255) / 256, 256, 0, stream>>>(src, dst, n);
  };
  cvt(kqv1w, wkqv1, DIM * 3 * DIM);
  cvt(kqv2w, wkqv2, DIM * 3 * DIM);
  cvt(proj1w, wproj1, DIM * DIM);
  cvt(proj2w, wproj2, DIM * DIM);
  cvt(fc1w1, wfc1_1, DIM * MLPH);
  cvt(fc2w1, wfc2_1, MLPH * DIM);
  cvt(fc1w2, wfc1_2, DIM * MLPH);
  cvt(fc2w2, wfc2_2, MLPH * DIM);

  ln_embed_kernel<<<dim3(NTOK / 32, BATCH), 256, 0, stream>>>(x1, lna1w, lna1b, xn1);
  ln_embed_kernel<<<dim3(NTOK / 32, BATCH), 256, 0, stream>>>(x2, lna2w, lna2b, xn2);

  gemm16<0><<<dim3(BN / 128, (3 * DIM) / 64), 256, 0, stream>>>(
      xn1, wkqv1, 3 * DIM, DIM, nullptr, nullptr, nullptr, nullptr, k1, q1, v1);
  gemm16<0><<<dim3(BN / 128, (3 * DIM) / 64), 256, 0, stream>>>(
      xn2, wkqv2, 3 * DIM, DIM, nullptr, nullptr, nullptr, nullptr, k2, q2, v2);

  attn_kernel<<<dim3(NTOK / 128, BATCH * HEADS), 256, 0, stream>>>(q1, k2, v2, a1);
  attn_kernel<<<dim3(NTOK / 128, BATCH * HEADS), 256, 0, stream>>>(q2, k1, v1, a2);

  gemm16<1><<<dim3(BN / 128, DIM / 64), 256, 0, stream>>>(
      a1, wproj1, DIM, DIM, proj1b, x1, pre1, nullptr, nullptr, nullptr, nullptr);
  gemm16<1><<<dim3(BN / 128, DIM / 64), 256, 0, stream>>>(
      a2, wproj2, DIM, DIM, proj2b, x2, pre2, nullptr, nullptr, nullptr, nullptr);

  ln_rows_kernel<<<BN / 32, 256, 0, stream>>>(pre1, ln1w, ln1b, xn1);
  ln_rows_kernel<<<BN / 32, 256, 0, stream>>>(pre2, ln2w, ln2b, xn2);

  gemm16<2><<<dim3(BN / 128, MLPH / 64), 256, 0, stream>>>(
      xn1, wfc1_1, MLPH, DIM, fc1b1, nullptr, nullptr, hid1, nullptr, nullptr, nullptr);
  gemm16<2><<<dim3(BN / 128, MLPH / 64), 256, 0, stream>>>(
      xn2, wfc1_2, MLPH, DIM, fc1b2, nullptr, nullptr, hid2, nullptr, nullptr, nullptr);

  gemm16<3><<<dim3(BN / 128, DIM / 64), 256, 0, stream>>>(
      hid1, wfc2_1, DIM, MLPH, fc2b1, pre1, out, nullptr, nullptr, nullptr, nullptr);
  gemm16<3><<<dim3(BN / 128, DIM / 64), 256, 0, stream>>>(
      hid2, wfc2_2, DIM, MLPH, fc2b2, pre2, out + (size_t)BATCH * DIM * NTOK,
      nullptr, nullptr, nullptr, nullptr);
}
